// TemporalGraphNetwork_22617297781311
// MI455X (gfx1250) — matrix operand order repaired and checked by reading, not compiled, not run on hardware
//
#include <hip/hip_runtime.h>
#include <hip/hip_bf16.h>
#include <math.h>

// ---------------------------------------------------------------------------
// Temporal Graph Network for MI455X (gfx1250), wave32 + WMMA f16.
// - All GEMMs (message MLP, GRU gates, GAT projections) use
//   v_wmma_f32_16x16x32_f16 with f16 weights converted once per launch.
// - M=32 row tiles per block: each B (weight) fragment is reused by two
//   M-tiles, halving L2 weight re-fetch traffic vs M=16.
// - x-feature gather staged with global_load_async_to_lds_b128 (+
//   s_wait_asynccnt), the CDNA5 async global->LDS path.
// - Scatter/softmax via float atomics + monotone-encoded u32 atomicMax.
// ---------------------------------------------------------------------------

typedef __attribute__((ext_vector_type(16))) _Float16 v16h;
typedef __attribute__((ext_vector_type(8)))  float    v8f;

#define TGN_N 100000
#define TGN_E 300000

// ----- monotone float <-> u32 encoding for atomic max over floats ---------
__device__ __forceinline__ unsigned fenc(float f) {
  unsigned u = __float_as_uint(f);
  return (u & 0x80000000u) ? ~u : (u | 0x80000000u);
}
__device__ __forceinline__ float fdec(unsigned e) {
  return __uint_as_float((e & 0x80000000u) ? (e & 0x7fffffffu) : ~e);
}

// ----- WMMA fragment loaders ----------------------------------------------
// A matrix 16xK (f16, row-major in LDS). Per ISA: lanes 0-15 row M=lane,
// VGPR pair p holds K = {2p, 2p+1} (+8 for lanes 16-31, +16 for p>=4).
__device__ __forceinline__ v16h load_a_frag_h(const _Float16* A, int ld,
                                              int lane, int k0) {
  const int row = lane & 15, hi = lane >> 4;
  const _Float16* r = A + row * ld;
  v16h a;
#pragma unroll
  for (int p = 0; p < 8; ++p) {
    int kb = k0 + ((p < 4) ? (hi * 8 + 2 * p) : (16 + hi * 8 + 2 * (p - 4)));
    a[2 * p]     = r[kb];
    a[2 * p + 1] = r[kb + 1];
  }
  return a;
}
// B matrix 32x16 taken as W^T where W is row-major [Nout][K] (f16 in global).
// B(k,n) = W[n0+n][k]; element s<8 carries K = k0+8*hi+s, s>=8 K = k0+16+8*hi+(s-8).
__device__ __forceinline__ v16h load_b_frag(const _Float16* W, int K, int n0,
                                            int lane, int k0) {
  const int n = lane & 15, hi = lane >> 4;
  const _Float16* wr = W + (size_t)(n0 + n) * K + k0 + hi * 8;
  v16h b;
#pragma unroll
  for (int j = 0; j < 16; ++j) b[j] = wr[(j < 8) ? j : (j + 8)];
  return b;
}
__device__ __forceinline__ v8f wmma16(v16h a, v16h b, v8f c) {
  return __builtin_amdgcn_wmma_f32_16x16x32_f16(false, a, false, b,
                                                (short)0, c, false, false);
}

// ----- utility kernels -----------------------------------------------------
__global__ void k_zero_f32(float* p, size_t n) {
  size_t i = (size_t)blockIdx.x * blockDim.x + threadIdx.x;
  if (i < n) p[i] = 0.0f;
}
__global__ void k_fill_u32(unsigned* p, unsigned v, size_t n) {
  size_t i = (size_t)blockIdx.x * blockDim.x + threadIdx.x;
  if (i < n) p[i] = v;
}
__global__ void k_f32_to_f16(const float* s, _Float16* d, size_t n) {
  size_t i = (size_t)blockIdx.x * blockDim.x + threadIdx.x;
  if (i < n) d[i] = (_Float16)s[i];
}
__global__ void k_timemax(const float* ts, unsigned* enc, int n) {
  int i = blockIdx.x * blockDim.x + threadIdx.x;
  if (i < n) atomicMax(enc, fenc(ts[i]));
}

// ----- message MLP: gather + 576->256 relu -> 256 + scatter, M=32 ---------
__global__ __launch_bounds__(256)
void k_messages(const float* __restrict__ node_memory,
                const float* __restrict__ xfeat,
                const float* __restrict__ last_upd,
                const int* __restrict__ src, const int* __restrict__ dst,
                const unsigned* __restrict__ tmax_enc,
                const float* __restrict__ tw1, const float* __restrict__ tb1,
                const float* __restrict__ tw2, const float* __restrict__ tb2,
                const _Float16* __restrict__ w1h, const float* __restrict__ b1,
                const _Float16* __restrict__ w2h, const float* __restrict__ b2,
                float* __restrict__ agg, float* __restrict__ counts) {
  __shared__ _Float16 Als[32][576];                  // 36.9 KB
  __shared__ _Float16 Hls[32][256];                  // 16.4 KB
  __shared__ __align__(16) float xstage[32 * 32];    //  4.0 KB (async dest)
  __shared__ int sS[32], sD[32];
  __shared__ float sdt[32];

  const int t = threadIdx.x;
  const int tile = blockIdx.x;
  if (t < 32) {
    int e = tile * 32 + t;
    int s = src[e], d = dst[e];
    sS[t] = s; sD[t] = d;
    float cur = fdec(*tmax_enc);
    sdt[t] = cur - fmaxf(last_upd[s], last_upd[d]);
    atomicAdd(&counts[d], 1.0f);
  }
  __syncthreads();

  // Async global->LDS gather of x[src] rows: 32 rows x 128B, one b128/lane.
  {
    int r = t >> 3, q = t & 7;
    const float* gp = xfeat + (size_t)sS[r] * 32 + q * 4;
    unsigned lo = (unsigned)(uintptr_t)(&xstage[r * 32 + q * 4]);
    asm volatile("global_load_async_to_lds_b128 %0, %1, off"
                 :: "v"(lo), "v"(gp) : "memory");
  }
  // VGPR-path staging of the two node-memory panels (f32 -> f16)
  for (int idx = t; idx < 32 * 256; idx += 256) {
    int r = idx >> 8, c = idx & 255;
    Als[r][c]       = (_Float16)node_memory[(size_t)sS[r] * 256 + c];
    Als[r][256 + c] = (_Float16)node_memory[(size_t)sD[r] * 256 + c];
  }
  // time encoding MLP: 1 -> 32 relu -> 32
  for (int idx = t; idx < 32 * 32; idx += 256) {
    int r = idx >> 5, o = idx & 31;
    float dt = sdt[r];
    float acc = tb2[o];
#pragma unroll
    for (int j = 0; j < 32; ++j) {
      float rj = fmaxf(dt * tw1[j] + tb1[j], 0.0f);
      acc += tw2[o * 32 + j] * rj;
    }
    Als[r][544 + o] = (_Float16)acc;
  }
  asm volatile("s_wait_asynccnt 0" ::: "memory");
  __syncthreads();
  for (int idx = t; idx < 32 * 32; idx += 256) {   // staged x -> f16
    int r = idx >> 5, c = idx & 31;
    Als[r][512 + c] = (_Float16)xstage[idx];
  }
  __syncthreads();

  const int wv = t >> 5, lane = t & 31, hi = lane >> 4, nn = lane & 15;
  for (int i = 0; i < 2; ++i) {                 // layer 1 (K = 576)
    int nt = wv * 2 + i;
    v8f acc0 = {}, acc1 = {};
    for (int kc = 0; kc < 18; ++kc) {
      v16h b  = load_b_frag(w1h, 576, nt * 16, lane, kc * 32);
      v16h a0 = load_a_frag_h(&Als[0][0],  576, lane, kc * 32);
      v16h a1 = load_a_frag_h(&Als[16][0], 576, lane, kc * 32);
      acc0 = wmma16(a0, b, acc0);
      acc1 = wmma16(a1, b, acc1);
    }
    int col = nt * 16 + nn;
    float bias = b1[col];
#pragma unroll
    for (int v = 0; v < 8; ++v) {
      int m = (hi ? 8 : 0) + v;
      Hls[m][col]      = (_Float16)fmaxf(acc0[v] + bias, 0.0f);
      Hls[16 + m][col] = (_Float16)fmaxf(acc1[v] + bias, 0.0f);
    }
  }
  __syncthreads();
  for (int i = 0; i < 2; ++i) {                 // layer 2 (K = 256) + scatter
    int nt = wv * 2 + i;
    v8f acc0 = {}, acc1 = {};
    for (int kc = 0; kc < 8; ++kc) {
      v16h b  = load_b_frag(w2h, 256, nt * 16, lane, kc * 32);
      v16h a0 = load_a_frag_h(&Hls[0][0],  256, lane, kc * 32);
      v16h a1 = load_a_frag_h(&Hls[16][0], 256, lane, kc * 32);
      acc0 = wmma16(a0, b, acc0);
      acc1 = wmma16(a1, b, acc1);
    }
    int col = nt * 16 + nn;
    float bias = b2[col];
#pragma unroll
    for (int v = 0; v < 8; ++v) {
      int m = (hi ? 8 : 0) + v;
      atomicAdd(&agg[(size_t)sD[m] * 256 + col],      acc0[v] + bias);
      atomicAdd(&agg[(size_t)sD[16 + m] * 256 + col], acc1[v] + bias);
    }
  }
}

// ----- GRU memory update (segment-mean + GRUCell), M=32 --------------------
__global__ __launch_bounds__(256)
void k_gru(const float* __restrict__ agg, const float* __restrict__ counts,
           const float* __restrict__ node_memory,
           const _Float16* __restrict__ wih16, const _Float16* __restrict__ whh16,
           const float* __restrict__ bih, const float* __restrict__ bhh,
           float* __restrict__ mem_out) {
  __shared__ _Float16 aggh[32][256];   // 16.4 KB
  __shared__ _Float16 memh[32][256];   // 16.4 KB
  __shared__ float scnt[32];
  const int t = threadIdx.x;
  const int base = blockIdx.x * 32;
  if (t < 32) scnt[t] = counts[base + t];
  __syncthreads();
  for (int idx = t; idx < 32 * 256; idx += 256) {
    int r = idx >> 8, c = idx & 255;
    aggh[r][c] =
        (_Float16)(agg[(size_t)(base + r) * 256 + c] / fmaxf(scnt[r], 1.0f));
    memh[r][c] = (_Float16)node_memory[(size_t)(base + r) * 256 + c];
  }
  __syncthreads();

  const int wv = t >> 5, lane = t & 31, hi = lane >> 4, nn = lane & 15;
  for (int sub = 0; sub < 2; ++sub) {
    int c0 = wv * 32 + sub * 16;    // state-column tile base
    v8f r0 = {}, r1 = {}, z0 = {}, z1 = {};
    for (int gate = 0; gate < 3; ++gate) {        // r, z, n
      int ncol0 = gate * 256 + c0;
      v8f ai0 = {}, ai1 = {}, ah0 = {}, ah1 = {};
      for (int kc = 0; kc < 8; ++kc) {
        v16h bi  = load_b_frag(wih16, 256, ncol0, lane, kc * 32);
        v16h bh  = load_b_frag(whh16, 256, ncol0, lane, kc * 32);
        v16h aa0 = load_a_frag_h(&aggh[0][0],  256, lane, kc * 32);
        v16h aa1 = load_a_frag_h(&aggh[16][0], 256, lane, kc * 32);
        v16h am0 = load_a_frag_h(&memh[0][0],  256, lane, kc * 32);
        v16h am1 = load_a_frag_h(&memh[16][0], 256, lane, kc * 32);
        ai0 = wmma16(aa0, bi, ai0);
        ai1 = wmma16(aa1, bi, ai1);
        ah0 = wmma16(am0, bh, ah0);
        ah1 = wmma16(am1, bh, ah1);
      }
      float bi_s = bih[ncol0 + nn], bh_s = bhh[ncol0 + nn];
      if (gate == 0) {
#pragma unroll
        for (int v = 0; v < 8; ++v) {
          r0[v] = 1.0f / (1.0f + expf(-(ai0[v] + bi_s + ah0[v] + bh_s)));
          r1[v] = 1.0f / (1.0f + expf(-(ai1[v] + bi_s + ah1[v] + bh_s)));
        }
      } else if (gate == 1) {
#pragma unroll
        for (int v = 0; v < 8; ++v) {
          z0[v] = 1.0f / (1.0f + expf(-(ai0[v] + bi_s + ah0[v] + bh_s)));
          z1[v] = 1.0f / (1.0f + expf(-(ai1[v] + bi_s + ah1[v] + bh_s)));
        }
      } else {
#pragma unroll
        for (int v = 0; v < 8; ++v) {
          int m = (hi ? 8 : 0) + v;
          int c = c0 + nn;
          float n0 = tanhf(ai0[v] + bi_s + r0[v] * (ah0[v] + bh_s));
          float n1 = tanhf(ai1[v] + bi_s + r1[v] * (ah1[v] + bh_s));
          float h0 = node_memory[(size_t)(base + m) * 256 + c];       // f32 h
          float h1 = node_memory[(size_t)(base + 16 + m) * 256 + c];
          float o0 = (1.0f - z0[v]) * n0 + z0[v] * h0;
          float o1 = (1.0f - z1[v]) * n1 + z1[v] * h1;
          mem_out[(size_t)(base + m) * 256 + c]      = scnt[m] > 0.0f ? o0 : h0;
          mem_out[(size_t)(base + 16 + m) * 256 + c] = scnt[16 + m] > 0.0f ? o1 : h1;
        }
      }
    }
  }
}

// ----- GAT projection: xh = h @ W^T + attention logits, M=32 ---------------
__global__ __launch_bounds__(256)
void k_gat_proj(const float* __restrict__ hin, const _Float16* __restrict__ w16,
                const float* __restrict__ a_src, const float* __restrict__ a_dst,
                float* __restrict__ xh, float* __restrict__ al_s,
                float* __restrict__ al_d) {
  __shared__ _Float16 hls[32][256];   // 16.4 KB
  __shared__ float    xls[32][256];   // 32.8 KB
  const int t = threadIdx.x;
  const int base = blockIdx.x * 32;
  for (int idx = t; idx < 32 * 256; idx += 256) {
    int r = idx >> 8, c = idx & 255;
    hls[r][c] = (_Float16)hin[(size_t)(base + r) * 256 + c];
  }
  __syncthreads();
  const int wv = t >> 5, lane = t & 31, hi = lane >> 4, nn = lane & 15;
  for (int i = 0; i < 2; ++i) {
    int nt = wv * 2 + i;
    v8f acc0 = {}, acc1 = {};
    for (int kc = 0; kc < 8; ++kc) {
      v16h b  = load_b_frag(w16, 256, nt * 16, lane, kc * 32);
      v16h a0 = load_a_frag_h(&hls[0][0],  256, lane, kc * 32);
      v16h a1 = load_a_frag_h(&hls[16][0], 256, lane, kc * 32);
      acc0 = wmma16(a0, b, acc0);
      acc1 = wmma16(a1, b, acc1);
    }
    int col = nt * 16 + nn;
#pragma unroll
    for (int v = 0; v < 8; ++v) {
      int m = (hi ? 8 : 0) + v;
      xls[m][col]      = acc0[v];
      xls[16 + m][col] = acc1[v];
      xh[(size_t)(base + m) * 256 + col]      = acc0[v];
      xh[(size_t)(base + 16 + m) * 256 + col] = acc1[v];
    }
  }
  __syncthreads();
  if (t < 128) {
    int m = t >> 2, h = t & 3;
    float ss = 0.0f, sd = 0.0f;
    for (int c = 0; c < 64; ++c) {
      float v = xls[m][h * 64 + c];
      ss += v * a_src[h * 64 + c];
      sd += v * a_dst[h * 64 + c];
    }
    al_s[(size_t)(base + m) * 4 + h] = ss;
    al_d[(size_t)(base + m) * 4 + h] = sd;
  }
}

// ----- GAT edge softmax (max, sum, weighted aggregate) ---------------------
__device__ __forceinline__ void edge_sd(int i, const int* src, const int* dst,
                                        int E_, int& s, int& d) {
  if (i < E_) { s = src[i]; d = dst[i]; } else { s = i - E_; d = s; }
}
__global__ void k_edge_max(const int* src, const int* dst, int E_, int tot,
                           const float* al_s, const float* al_d,
                           unsigned* emax) {
  int i = blockIdx.x * blockDim.x + threadIdx.x;
  if (i >= tot) return;
  int s, d; edge_sd(i, src, dst, E_, s, d);
#pragma unroll
  for (int h = 0; h < 4; ++h) {
    float e = al_s[(size_t)s * 4 + h] + al_d[(size_t)d * 4 + h];
    e = e > 0.0f ? e : 0.2f * e;
    atomicMax(&emax[(size_t)d * 4 + h], fenc(e));
  }
}
__global__ void k_edge_sum(const int* src, const int* dst, int E_, int tot,
                           const float* al_s, const float* al_d,
                           const unsigned* emax, float* denom) {
  int i = blockIdx.x * blockDim.x + threadIdx.x;
  if (i >= tot) return;
  int s, d; edge_sd(i, src, dst, E_, s, d);
#pragma unroll
  for (int h = 0; h < 4; ++h) {
    float e = al_s[(size_t)s * 4 + h] + al_d[(size_t)d * 4 + h];
    e = e > 0.0f ? e : 0.2f * e;
    atomicAdd(&denom[(size_t)d * 4 + h], expf(e - fdec(emax[(size_t)d * 4 + h])));
  }
}
__global__ void k_edge_out(const int* src, const int* dst, int E_, int tot4,
                           const float* al_s, const float* al_d,
                           const unsigned* emax, const float* denom,
                           const float* __restrict__ xh, float* __restrict__ out) {
  int g = blockIdx.x * blockDim.x + threadIdx.x;
  if (g >= tot4) return;
  int i = g >> 2, h = g & 3;
  int s, d; edge_sd(i, src, dst, E_, s, d);
  float e = al_s[(size_t)s * 4 + h] + al_d[(size_t)d * 4 + h];
  e = e > 0.0f ? e : 0.2f * e;
  float alpha = expf(e - fdec(emax[(size_t)d * 4 + h])) / denom[(size_t)d * 4 + h];
  const float* xs = xh + (size_t)s * 256 + h * 64;
  float* od = out + (size_t)d * 256 + h * 64;
  for (int c = 0; c < 64; ++c) atomicAdd(&od[c], alpha * xs[c]);
}
__global__ void k_gat_finalize(float* out, const float* bias, size_t n) {
  size_t i = (size_t)blockIdx.x * blockDim.x + threadIdx.x;
  if (i >= n) return;
  float v = out[i] + bias[i & 255];
  out[i] = v > 0.0f ? v : (expf(v) - 1.0f);   // ELU
}

// ----- global mean pool + classifier ---------------------------------------
__global__ void k_pool(const float* h, float* pooled, size_t n) {
  size_t i = (size_t)blockIdx.x * blockDim.x + threadIdx.x;
  if (i < n) atomicAdd(&pooled[i & 255], h[i]);
}
__global__ __launch_bounds__(128)
void k_cls(const float* pooled, const float* w1, const float* b1,
           const float* w2, const float* b2, float* outp, float invn) {
  __shared__ float g[256];
  __shared__ float hid[128];
  int t = threadIdx.x;
  for (int c = t; c < 256; c += 128) g[c] = pooled[c] * invn;
  __syncthreads();
  float acc = b1[t];
  for (int c = 0; c < 256; ++c) acc += g[c] * w1[t * 256 + c];
  hid[t] = fmaxf(acc, 0.0f);
  __syncthreads();
  if (t < 2) {
    float a = b2[t];
    for (int j = 0; j < 128; ++j) a += hid[j] * w2[t * 128 + j];
    outp[t] = a;
  }
}

// ---------------------------------------------------------------------------
extern "C" void kernel_launch(void* const* d_in, const int* in_sizes, int n_in,
                              void* d_out, int out_size, void* d_ws,
                              size_t ws_size, hipStream_t stream) {
  (void)in_sizes; (void)n_in; (void)out_size; (void)ws_size;
  const int N_ = TGN_N, E_ = TGN_E;
  const size_t NB = (size_t)N_ * 256;

  const float* x        = (const float*)d_in[0];
  const float* ts       = (const float*)d_in[1];
  const float* node_mem = (const float*)d_in[2];
  const float* last_upd = (const float*)d_in[3];
  const float* msg_w1   = (const float*)d_in[4];
  const float* msg_b1   = (const float*)d_in[5];
  const float* msg_w2   = (const float*)d_in[6];
  const float* msg_b2   = (const float*)d_in[7];
  const float* time_w1  = (const float*)d_in[8];
  const float* time_b1  = (const float*)d_in[9];
  const float* time_w2  = (const float*)d_in[10];
  const float* time_b2  = (const float*)d_in[11];
  const float* gru_wih  = (const float*)d_in[12];
  const float* gru_whh  = (const float*)d_in[13];
  const float* gru_bih  = (const float*)d_in[14];
  const float* gru_bhh  = (const float*)d_in[15];
  const float* gat1_w   = (const float*)d_in[16];
  const float* gat1_as  = (const float*)d_in[17];
  const float* gat1_ad  = (const float*)d_in[18];
  const float* gat1_b   = (const float*)d_in[19];
  const float* gat2_w   = (const float*)d_in[20];
  const float* gat2_as  = (const float*)d_in[21];
  const float* gat2_ad  = (const float*)d_in[22];
  const float* gat2_b   = (const float*)d_in[23];
  const float* cls_w1   = (const float*)d_in[24];
  const float* cls_b1   = (const float*)d_in[25];
  const float* cls_w2   = (const float*)d_in[26];
  const float* cls_b2   = (const float*)d_in[27];
  const int*   eidx     = (const int*)d_in[28];
  const int*   src      = eidx;
  const int*   dst      = eidx + E_;

  // workspace layout (~210 MB): two big N*256 f32 buffers + small arrays
  char* ws = (char*)d_ws;
  size_t off = 0;
  auto take = [&](size_t bytes) {
    size_t o = off; off += (bytes + 255) & ~(size_t)255; return o;
  };
  unsigned*  tmax   = (unsigned*)(ws + take(4));
  float*     counts = (float*)(ws + take((size_t)N_ * 4));
  float*     pooled = (float*)(ws + take(256 * 4));
  float*     al_s   = (float*)(ws + take((size_t)N_ * 4 * 4));
  float*     al_d   = (float*)(ws + take((size_t)N_ * 4 * 4));
  unsigned*  emax   = (unsigned*)(ws + take((size_t)N_ * 4 * 4));
  float*     denom  = (float*)(ws + take((size_t)N_ * 4 * 4));
  _Float16*  w16_m1 = (_Float16*)(ws + take((size_t)256 * 576 * 2));
  _Float16*  w16_m2 = (_Float16*)(ws + take((size_t)256 * 256 * 2));
  _Float16*  w16_ih = (_Float16*)(ws + take((size_t)768 * 256 * 2));
  _Float16*  w16_hh = (_Float16*)(ws + take((size_t)768 * 256 * 2));
  _Float16*  w16_g1 = (_Float16*)(ws + take((size_t)256 * 256 * 2));
  _Float16*  w16_g2 = (_Float16*)(ws + take((size_t)256 * 256 * 2));
  float*     bufA   = (float*)(ws + take(NB * 4));  // agg, later xh
  float*     bufB   = (float*)(ws + take(NB * 4));  // mem, later h1/h2

  const int B = 256;
  auto blk  = [&](size_t n) { return (unsigned)((n + B - 1) / B); };

  // init
  k_fill_u32<<<1, 1, 0, stream>>>(tmax, 0u, 1);
  k_zero_f32<<<blk(N_), B, 0, stream>>>(counts, N_);
  k_zero_f32<<<blk(NB), B, 0, stream>>>(bufA, NB);
  k_zero_f32<<<1, B, 0, stream>>>(pooled, 256);

  // f32 -> f16 weight conversion (tiny, L2-resident afterwards)
  k_f32_to_f16<<<blk(256 * 576), B, 0, stream>>>(msg_w1, w16_m1, 256 * 576);
  k_f32_to_f16<<<blk(256 * 256), B, 0, stream>>>(msg_w2, w16_m2, 256 * 256);
  k_f32_to_f16<<<blk(768 * 256), B, 0, stream>>>(gru_wih, w16_ih, 768 * 256);
  k_f32_to_f16<<<blk(768 * 256), B, 0, stream>>>(gru_whh, w16_hh, 768 * 256);
  k_f32_to_f16<<<blk(256 * 256), B, 0, stream>>>(gat1_w, w16_g1, 256 * 256);
  k_f32_to_f16<<<blk(256 * 256), B, 0, stream>>>(gat2_w, w16_g2, 256 * 256);

  // current_time = max(timestamps)
  k_timemax<<<blk(E_), B, 0, stream>>>(ts, tmax, E_);

  // messages: WMMA 576->256 relu -> 256, fused scatter into bufA(agg)
  k_messages<<<E_ / 32, B, 0, stream>>>(
      node_mem, x, last_upd, src, dst, tmax,
      time_w1, time_b1, time_w2, time_b2,
      w16_m1, msg_b1, w16_m2, msg_b2, bufA, counts);

  // GRU -> bufB(mem)
  k_gru<<<N_ / 32, B, 0, stream>>>(bufA, counts, node_mem, w16_ih, w16_hh,
                                   gru_bih, gru_bhh, bufB);

  const int tot  = E_ + N_;      // edges + self-loops
  const int tot4 = tot * 4;

  // ----- GAT layer 1: in=bufB(mem), xh=bufA, out/h1=bufB -----
  k_gat_proj<<<N_ / 32, B, 0, stream>>>(bufB, w16_g1, gat1_as, gat1_ad,
                                        bufA, al_s, al_d);
  k_zero_f32<<<blk(NB), B, 0, stream>>>(bufB, NB);
  k_fill_u32<<<blk((size_t)N_ * 4), B, 0, stream>>>(emax, 0u, (size_t)N_ * 4);
  k_zero_f32<<<blk((size_t)N_ * 4), B, 0, stream>>>(denom, (size_t)N_ * 4);
  k_edge_max<<<blk(tot), B, 0, stream>>>(src, dst, E_, tot, al_s, al_d, emax);
  k_edge_sum<<<blk(tot), B, 0, stream>>>(src, dst, E_, tot, al_s, al_d, emax, denom);
  k_edge_out<<<blk(tot4), B, 0, stream>>>(src, dst, E_, tot4, al_s, al_d,
                                          emax, denom, bufA, bufB);
  k_gat_finalize<<<blk(NB), B, 0, stream>>>(bufB, gat1_b, NB);

  // ----- GAT layer 2: in=bufB(h1), xh=bufA, out/h2=bufB -----
  k_gat_proj<<<N_ / 32, B, 0, stream>>>(bufB, w16_g2, gat2_as, gat2_ad,
                                        bufA, al_s, al_d);
  k_zero_f32<<<blk(NB), B, 0, stream>>>(bufB, NB);
  k_fill_u32<<<blk((size_t)N_ * 4), B, 0, stream>>>(emax, 0u, (size_t)N_ * 4);
  k_zero_f32<<<blk((size_t)N_ * 4), B, 0, stream>>>(denom, (size_t)N_ * 4);
  k_edge_max<<<blk(tot), B, 0, stream>>>(src, dst, E_, tot, al_s, al_d, emax);
  k_edge_sum<<<blk(tot), B, 0, stream>>>(src, dst, E_, tot, al_s, al_d, emax, denom);
  k_edge_out<<<blk(tot4), B, 0, stream>>>(src, dst, E_, tot4, al_s, al_d,
                                          emax, denom, bufA, bufB);
  k_gat_finalize<<<blk(NB), B, 0, stream>>>(bufB, gat2_b, NB);

  // pool + classifier
  k_pool<<<blk(NB), B, 0, stream>>>(bufB, pooled, NB);
  k_cls<<<1, 128, 0, stream>>>(pooled, cls_w1, cls_b1, cls_w2, cls_b2,
                               (float*)d_out, 1.0f / (float)N_);
}
